// VectorQuantizer_43679817401100
// MI455X (gfx1250) — matrix operand order repaired and checked by reading, not compiled, not run on hardware
//
#include <hip/hip_runtime.h>
#include <cstdint>

// ---------------- CDNA5 (gfx1250, wave32) fused VQ-VAE quantizer ----------------
// dist(n,c) = ||z||^2 + ||e_c||^2 - 2 z.e ; argmin needs only (||e||^2 - 2 z.e).
// GEMM: v_wmma_f32_16x16x32_f16 with split-f16 (hi+lo) 3-multiply emulation (~f32 fidelity).
// Codebook pre-swizzled once into ws in B-tile layout; per-block staging uses
// global_load_async_to_lds_b128 double-buffered under s_wait_asynccnt.

typedef __attribute__((ext_vector_type(16))) _Float16 v16h;
typedef __attribute__((ext_vector_type(8)))  float    v8f;
typedef __attribute__((ext_vector_type(4)))  unsigned int u32x4;
typedef __attribute__((ext_vector_type(4)))  float    f32x4;

union V16H { v16h v; _Float16 h[16]; u32x4 q[2]; };

#define NB       16
#define DDIM     64
#define TT       16384
#define NCODES   1024
#define NTOT     (NB * TT)          /* 262144 z-vectors */
#define ROWS     128                /* z rows per block */
#define G1       (NTOT / ROWS)      /* 2048 blocks */
#define ZQ_ELEMS (NB * DDIM * TT)   /* 16777216 */
#define ROW_Q    9                  /* u32x4 per padded code row (36 dwords) */
#define HALF_Q   (64 * ROW_Q)       /* 576 u32x4 per (hi|lo) half chunk */
#define CHUNK_Q  (2 * HALF_Q)       /* 1152 u32x4 per 64-code chunk */
#define WS_CB_F  4096               /* float offset of packed codebook in ws */

// ws (floats): [0,1024) csq | [1024,2048) used | [2048,4096) partials | [4096,..) packed cb

__global__ __launch_bounds__(256) void vq_init(const float* __restrict__ cb,
                                               float* __restrict__ ws,
                                               unsigned* __restrict__ cbp) {
    int c = blockIdx.x * 256 + threadIdx.x;   // grid=8 -> c in [0,2048)
    if (c < NCODES) {
        int cc = c >> 6, r = c & 63;
        unsigned hbase = (unsigned)(cc * CHUNK_Q + r * ROW_Q) * 4u;  // uint index
        unsigned lbase = hbase + HALF_Q * 4u;
        float s = 0.f;
        #pragma unroll
        for (int du = 0; du < 32; ++du) {
            float g0 = cb[c * DDIM + 2 * du], g1 = cb[c * DDIM + 2 * du + 1];
            s += g0 * g0 + g1 * g1;
            _Float16 h0 = (_Float16)g0, h1 = (_Float16)g1;
            union { _Float16 h[2]; unsigned u; } ph, pl;
            ph.h[0] = h0;                       ph.h[1] = h1;
            pl.h[0] = (_Float16)(g0 - (float)h0); pl.h[1] = (_Float16)(g1 - (float)h1);
            cbp[hbase + du] = ph.u;
            cbp[lbase + du] = pl.u;
        }
        ws[c] = s;              // ||e_c||^2
        ws[NCODES + c] = 0.f;   // used flag
    }
    if (c < G1) ws[2048 + c] = 0.f;  // partials
}

__global__ __launch_bounds__(256) void vq_argmin(const float* __restrict__ ze,
                                                 const u32x4* __restrict__ cbpk,
                                                 const float* __restrict__ csqg,
                                                 float* __restrict__ idx_out,
                                                 float* __restrict__ ws) {
    __shared__ union Sh {
        float ztile[DDIM][ROWS];                          // 32 KB (start only)
        u32x4 buf[2][CHUNK_Q];                            // 36.9 KB (main loop)
        struct { float v[ROWS * 16]; int i[ROWS * 16]; } red;  // 16 KB (end only)
    } sh;
    __shared__ float rsum[256];

    const int tid = threadIdx.x;
    const int w = tid >> 5, L = tid & 31, lr = L & 15, hf = L >> 4;
    const int n0 = blockIdx.x * ROWS;
    const int bb = n0 >> 14, t0 = n0 & (TT - 1);

    // ---- async-DMA the z tile (f32, d-major) into LDS: 8x b128 per thread ----
    const float* zbase = ze + (size_t)(bb << 6) * TT + t0;
    #pragma unroll
    for (int i = 0; i < 8; ++i) {
        int j = i * 256 + tid;             // 2048 u32x4
        int d = j >> 5, n4 = j & 31;
        unsigned lds  = (unsigned)(uintptr_t)&sh.ztile[d][n4 * 4];
        unsigned goff = (unsigned)(d * TT + n4 * 4) * 4u;
        asm volatile("global_load_async_to_lds_b128 %0, %1, %2"
                     :: "v"(lds), "v"(goff), "s"(zbase) : "memory");
    }
    asm volatile("s_wait_asynccnt 0x0" ::: "memory");
    __syncthreads();

    // ---- build A (16x64 split f16) + per-row ||z||^2 while ztile alive ----
    // A layout (16-bit A 16x32): lanes0-15: M=lane, h[0..7]=K0..7, h[8..15]=K16..23
    //                            lanes16-31: M=lane-16, K8..15 / K24..31
    V16H ah[2], al[2];
    const int nloc = w * 16 + lr;
    #pragma unroll
    for (int kc = 0; kc < 2; ++kc) {
        #pragma unroll
        for (int j = 0; j < 8; ++j) {
            int d1 = kc * 32 + hf * 8 + j;
            float x = sh.ztile[d1][nloc];
            _Float16 h = (_Float16)x;
            ah[kc].h[j] = h;  al[kc].h[j] = (_Float16)(x - (float)h);
            float x2 = sh.ztile[d1 + 16][nloc];
            _Float16 h2 = (_Float16)x2;
            ah[kc].h[8 + j] = h2;  al[kc].h[8 + j] = (_Float16)(x2 - (float)h2);
        }
    }
    float myzsq = 0.f;
    if (tid < ROWS) {
        #pragma unroll
        for (int d = 0; d < DDIM; ++d) { float x = sh.ztile[d][tid]; myzsq += x * x; }
    }
    __syncthreads();   // everyone done reading ztile -> buffers may overwrite it

    // ---- issue chunk 0 into buf[0] (5x b128 per thread, uniform issue count) ----
    {
        const u32x4* src = cbpk;
        #pragma unroll
        for (int i = 0; i < 5; ++i) {
            int j = i * 256 + tid; if (j >= CHUNK_Q) j = CHUNK_Q - 1;  // idempotent tail
            unsigned lds  = (unsigned)(uintptr_t)&sh.buf[0][j];
            unsigned goff = (unsigned)j * 16u;
            asm volatile("global_load_async_to_lds_b128 %0, %1, %2"
                         :: "v"(lds), "v"(goff), "s"(src) : "memory");
        }
    }

    float best[8]; int bidx[8];
    #pragma unroll
    for (int i = 0; i < 8; ++i) { best[i] = 3.4e38f; bidx[i] = 0; }

    // ---- sweep 1024 codes: 16 chunks of 64, double-buffered async staging ----
    for (int cc = 0; cc < 16; ++cc) {
        if (cc < 15) {
            __syncthreads();   // all waves consumed chunk cc-1 -> buffer (cc+1)&1 free
            const u32x4* src = cbpk + (cc + 1) * CHUNK_Q;
            #pragma unroll
            for (int i = 0; i < 5; ++i) {
                int j = i * 256 + tid; if (j >= CHUNK_Q) j = CHUNK_Q - 1;
                unsigned lds  = (unsigned)(uintptr_t)&sh.buf[(cc + 1) & 1][j];
                unsigned goff = (unsigned)j * 16u;
                asm volatile("global_load_async_to_lds_b128 %0, %1, %2"
                             :: "v"(lds), "v"(goff), "s"(src) : "memory");
            }
            asm volatile("s_wait_asynccnt 0x5" ::: "memory");  // chunk cc landed (mine)
        } else {
            __syncthreads();
            asm volatile("s_wait_asynccnt 0x0" ::: "memory");
        }
        __syncthreads();       // all waves' chunk-cc portions landed

        const u32x4* bp = sh.buf[cc & 1];
        #pragma unroll
        for (int sub = 0; sub < 4; ++sub) {
            const int cb0 = cc * 64 + sub * 16;
            const int lc0 = sub * 16;
            // B layout (Kx16): lanes0-15: N=lane, K0..7,K16..23 ; lanes16-31: N=lane-16, K8..15,K24..31
            //  -> 2 u32x4 per v16h  (ds_load_b128, stride 9 q = conflict-free)
            V16H bh[2], bl[2];
            #pragma unroll
            for (int kc = 0; kc < 2; ++kc) {
                int base = (lc0 + lr) * ROW_Q + kc * 4 + hf;
                bh[kc].q[0] = bp[base];           bh[kc].q[1] = bp[base + 2];
                bl[kc].q[0] = bp[HALF_Q + base];  bl[kc].q[1] = bp[HALF_Q + base + 2];
            }

            v8f acc = {0.f, 0.f, 0.f, 0.f, 0.f, 0.f, 0.f, 0.f};
            #pragma unroll
            for (int kc = 0; kc < 2; ++kc) {
                acc = __builtin_amdgcn_wmma_f32_16x16x32_f16(false, ah[kc].v, false, bh[kc].v,
                                                             (short)0, acc, false, false);
                acc = __builtin_amdgcn_wmma_f32_16x16x32_f16(false, ah[kc].v, false, bl[kc].v,
                                                             (short)0, acc, false, false);
                acc = __builtin_amdgcn_wmma_f32_16x16x32_f16(false, al[kc].v, false, bh[kc].v,
                                                             (short)0, acc, false, false);
            }

            const float cs   = csqg[cb0 + lr];   // ||e||^2, L2/L0-resident
            const int   code = cb0 + lr;
            #pragma unroll
            for (int i = 0; i < 8; ++i) {
                float v = cs - 2.f * acc[i];
                if (v < best[i]) { best[i] = v; bidx[i] = code; }
            }
        }
    }

    // ---- cross-lane argmin + deterministic loss partials ----
    __syncthreads();   // done with buffers; alias as reduction scratch
    #pragma unroll
    for (int i = 0; i < 8; ++i) {
        int row = w * 16 + hf * 8 + i;   // C layout: vgpr i -> M=i (+8 upper lanes)
        sh.red.v[row * 16 + lr] = best[i];
        sh.red.i[row * 16 + lr] = bidx[i];
    }
    __syncthreads();

    float diff = 0.f;
    if (tid < ROWS) {
        float bv = sh.red.v[tid * 16];
        int   bi = sh.red.i[tid * 16];
        #pragma unroll
        for (int j = 1; j < 16; ++j) {
            float v = sh.red.v[tid * 16 + j];
            int   c = sh.red.i[tid * 16 + j];
            if (v < bv || (v == bv && c < bi)) { bv = v; bi = c; }  // first-min tiebreak
        }
        diff = myzsq + bv;                  // ||z - e_best||^2 for this row
        idx_out[n0 + tid] = (float)bi;
        ws[NCODES + bi] = 1.0f;             // idempotent used-flag
    }
    rsum[tid] = diff;
    __syncthreads();
    for (int s = 128; s > 0; s >>= 1) {
        if (tid < s) rsum[tid] += rsum[tid + s];
        __syncthreads();
    }
    if (tid == 0) ws[2048 + blockIdx.x] = rsum[0];
}

__global__ __launch_bounds__(256) void vq_gather(const float* __restrict__ cb,
                                                 const float* __restrict__ idxf,
                                                 f32x4* __restrict__ zq) {
    int e4 = blockIdx.x * 256 + threadIdx.x;   // 4 consecutive t per thread
    int t4 = (e4 & 4095) << 2;
    int d  = (e4 >> 12) & 63;
    int b  = e4 >> 18;
    f32x4 iv = *(const f32x4*)(idxf + b * TT + t4);   // 16B-aligned
    f32x4 o;
    #pragma unroll
    for (int k = 0; k < 4; ++k) o[k] = cb[((int)iv[k]) * DDIM + d];  // cb L2-resident
    zq[e4] = o;                                        // coalesced b128 store
}

__global__ __launch_bounds__(1024) void vq_final(const float* __restrict__ ws,
                                                 float* __restrict__ sc) {
    __shared__ float sv[1024], su[1024];
    int tid = threadIdx.x;
    sv[tid] = ws[2048 + tid] + ws[2048 + 1024 + tid];
    su[tid] = ws[NCODES + tid];
    __syncthreads();
    for (int s = 512; s > 0; s >>= 1) {
        if (tid < s) { sv[tid] += sv[tid + s]; su[tid] += su[tid + s]; }
        __syncthreads();
    }
    if (tid == 0) {
        float mean = sv[0] / (float)ZQ_ELEMS;
        sc[0] = 1.25f * mean;            // vq_loss
        sc[1] = mean;                    // codebook_loss
        sc[2] = mean;                    // commitment_loss
        sc[3] = su[0] / (float)NCODES;   // utilization
    }
}

extern "C" void kernel_launch(void* const* d_in, const int* in_sizes, int n_in,
                              void* d_out, int out_size, void* d_ws, size_t ws_size,
                              hipStream_t stream) {
    (void)in_sizes; (void)n_in; (void)out_size; (void)ws_size;
    const float* ze = (const float*)d_in[0];   // (16, 64, 16384) f32
    const float* cb = (const float*)d_in[1];   // (1024, 64) f32
    float* out  = (float*)d_out;
    float* ws   = (float*)d_ws;
    float* zq   = out;                         // 16777216
    float* idxf = out + ZQ_ELEMS;              // 262144 (indices as float)
    float* sc   = idxf + NTOT;                 // 4 scalars
    unsigned* cbp = (unsigned*)(ws + WS_CB_F); // packed split-f16 codebook (288 KB)

    vq_init  <<<8,              256, 0, stream>>>(cb, ws, cbp);
    vq_argmin<<<G1,             256, 0, stream>>>(ze, (const u32x4*)cbp, ws, idxf, ws);
    vq_gather<<<ZQ_ELEMS/1024,  256, 0, stream>>>(cb, idxf, (f32x4*)zq);
    vq_final <<<1,             1024, 0, stream>>>(ws, sc);
}
